// SemanticGraphBuilder_30794915512742
// MI455X (gfx1250) — matrix operand order repaired and checked by reading, not compiled, not run on hardware
//
#include <hip/hip_runtime.h>
#include <hip/hip_bf16.h>
#include <math.h>

#define Bb 4
#define Nn 256
#define Dd 256

typedef __attribute__((ext_vector_type(16))) __bf16 v16bf;
typedef __attribute__((ext_vector_type(8)))  float  v8f;

__device__ __forceinline__ unsigned short f2bf(float f) {
  unsigned u = __float_as_uint(f);
  u += 0x7FFFu + ((u >> 16) & 1u);     // round-to-nearest-even
  return (unsigned short)(u >> 16);
}

// ---------------------------------------------------------------------------
// Prep: (1) srcb = shared@Wa + b1, tgt = shared@Wb  (factored out of pair loop)
//       (2) Wc/Wd -> bf16, pre-swizzled into WMMA B-fragment lane layout:
//           flat = (((ht*8+ks)*32 + lane)*16 + e), element (lane,e) holds
//           W[d = ks*32 + (lane>>4)*8 + (e&7) + (e>>3)*16][h = ht*16 + (lane&15)]
// ---------------------------------------------------------------------------
__global__ __launch_bounds__(256) void sgb_prep(
    const float* __restrict__ sh, const float* __restrict__ W1,
    const float* __restrict__ b1,
    float* __restrict__ srcb, float* __restrict__ tgt,
    unsigned short* __restrict__ WcS, unsigned short* __restrict__ WdS)
{
  const int blk = blockIdx.x;
  const int tid = threadIdx.x;
  if (blk < Bb * Nn) {
    __shared__ float srow[Dd];
    const float* sp = sh + (size_t)blk * Dd;
    srow[tid] = sp[tid];
    __syncthreads();
    float a0 = 0.f, a1 = 0.f;
    const float* wa = W1 + tid;              // Wa[d][tid], row-major (4D,D)
    const float* wb = W1 + Dd * Dd + tid;    // Wb[d][tid]
#pragma unroll 4
    for (int d = 0; d < Dd; ++d) {
      const float sv = srow[d];
      a0 = fmaf(sv, wa[d * Dd], a0);
      a1 = fmaf(sv, wb[d * Dd], a1);
    }
    srcb[(size_t)blk * Dd + tid] = a0 + b1[tid];
    tgt [(size_t)blk * Dd + tid] = a1;
  } else {
    const int idx0 = (blk - Bb * Nn) * 256 + tid;   // 0..32767
#pragma unroll
    for (int t = 0; t < 2; ++t) {
      const int idx = idx0 + t * 32768;             // 0..65535
      const int e  = idx & 15;
      const int l  = (idx >> 4) & 31;
      const int ks = (idx >> 9) & 7;
      const int ht = idx >> 12;
      const int d  = ks * 32 + (l >> 4) * 8 + (e & 7) + (e >> 3) * 16;
      const int h  = ht * 16 + (l & 15);
      WcS[idx] = f2bf(W1[(2 * Dd + d) * Dd + h]);   // Wc = W1[2D:3D]
      WdS[idx] = f2bf(W1[(3 * Dd + d) * Dd + h]);   // Wd = W1[3D:4D]
    }
  }
}

// ---------------------------------------------------------------------------
// Main: one workgroup (8 waves) per (b,i).  Per 128-row j-slab: build
// diff/prod bf16 in LDS, each wave owns a 16-row j-tile, caches full-K A
// fragments in registers, sweeps 16 h-tiles with TWO independent WMMA
// accumulator chains (diff@Wc and prod@Wd), fuses SiLU + W2-dot, then
// masked softmax.
// ---------------------------------------------------------------------------
__global__ __launch_bounds__(256) void sgb_main(
    const float* __restrict__ sh,
    const float* __restrict__ W2,
    const float* __restrict__ b2,
    const float* __restrict__ srcb,
    const float* __restrict__ tgt,
    const unsigned short* __restrict__ WcS,
    const unsigned short* __restrict__ WdS,
    float* __restrict__ out)
{
  constexpr int LDA = Dd + 8;                          // 264 halves: conflict-free
  __shared__ alignas(16) unsigned short diff_s[128 * LDA];
  __shared__ alignas(16) unsigned short prod_s[128 * LDA];
  __shared__ alignas(16) float si_s[Dd];
  __shared__ float srcb_s[Dd];
  __shared__ float w2_s[Dd];
  __shared__ float scores_s[Nn];
  __shared__ float red_s[8];

  const int i    = blockIdx.x;
  const int b    = blockIdx.y;
  const int tid  = threadIdx.x;
  const int lane = tid & 31;
  const int wave = tid >> 5;
  const int half = lane >> 4;
  const int l15  = lane & 15;

  const size_t bi = (size_t)(b * Nn + i);
  si_s[tid]   = sh[bi * Dd + tid];
  srcb_s[tid] = srcb[bi * Dd + tid];
  w2_s[tid]   = W2[tid];
  const float b2v = b2[0];
  __syncthreads();

  union U16 { uint4 u[2]; v16bf v; };

  for (int slab = 0; slab < 2; ++slab) {
    const int j0 = slab * 128;

    // ---- build |si-sj| and si*sj slabs in LDS (bf16) ----
    for (int jr = wave; jr < 128; jr += 8) {
      const float4* sj4 = (const float4*)(sh + (size_t)(b * Nn + j0 + jr) * Dd);
      const float4* si4 = (const float4*)si_s;
      unsigned short* dr = diff_s + jr * LDA;
      unsigned short* pr = prod_s + jr * LDA;
      for (int q = lane; q < Dd / 4; q += 32) {
        const float4 a = si4[q], c = sj4[q];
        uint2 dv, pv;
        dv.x = (unsigned)f2bf(fabsf(a.x - c.x)) | ((unsigned)f2bf(fabsf(a.y - c.y)) << 16);
        dv.y = (unsigned)f2bf(fabsf(a.z - c.z)) | ((unsigned)f2bf(fabsf(a.w - c.w)) << 16);
        pv.x = (unsigned)f2bf(a.x * c.x)        | ((unsigned)f2bf(a.y * c.y) << 16);
        pv.y = (unsigned)f2bf(a.z * c.z)        | ((unsigned)f2bf(a.w * c.w) << 16);
        *(uint2*)(dr + q * 4) = dv;
        *(uint2*)(pr + q * 4) = pv;
      }
    }
    __syncthreads();

    // ---- cache full-K A fragments for this wave's 16-row j-tile ----
    // A layout (16-bit, 16x32): lane M = l15; e<8 -> K = ks*32+half*8+e,
    //                           e>=8 -> K = ks*32+16+half*8+(e-8)
    const int jrow = wave * 16 + l15;
    v16bf AD[8], AP[8];
#pragma unroll
    for (int ks = 0; ks < 8; ++ks) {
      const int c0 = ks * 32 + half * 8;
      U16 ua, up;
      ua.u[0] = *(const uint4*)(diff_s + jrow * LDA + c0);
      ua.u[1] = *(const uint4*)(diff_s + jrow * LDA + c0 + 16);
      up.u[0] = *(const uint4*)(prod_s + jrow * LDA + c0);
      up.u[1] = *(const uint4*)(prod_s + jrow * LDA + c0 + 16);
      AD[ks] = ua.v;
      AP[ks] = up.v;
    }

    float partial[8] = {0.f, 0.f, 0.f, 0.f, 0.f, 0.f, 0.f, 0.f};
    for (int ht = 0; ht < 16; ++ht) {
      // two independent accumulator chains -> consecutive WMMAs have no RAW
      v8f accC = {0.f, 0.f, 0.f, 0.f, 0.f, 0.f, 0.f, 0.f};
      v8f accD = {0.f, 0.f, 0.f, 0.f, 0.f, 0.f, 0.f, 0.f};
#pragma unroll
      for (int ks = 0; ks < 8; ++ks) {
        U16 bc, bd;
        const uint4* pc = (const uint4*)(WcS + (size_t)(((ht * 8 + ks) * 32 + lane) * 16));
        const uint4* pd = (const uint4*)(WdS + (size_t)(((ht * 8 + ks) * 32 + lane) * 16));
        bc.u[0] = pc[0]; bc.u[1] = pc[1];
        bd.u[0] = pd[0]; bd.u[1] = pd[1];
        accC = __builtin_amdgcn_wmma_f32_16x16x32_bf16(false, AD[ks], false, bc.v,
                                                       (short)0, accC, false, false);
        accD = __builtin_amdgcn_wmma_f32_16x16x32_bf16(false, AP[ks], false, bd.v,
                                                       (short)0, accD, false, false);
      }
      // ---- fused epilogue: h = accC+accD + srcb[h] + tgt[j,h]; silu; dot W2 ----
      // C/D layout: element r, lane -> M = r + 8*half (j row), N = l15 (h col)
      const int   hcol = ht * 16 + l15;
      const float w2v  = w2_s[hcol];
      const float sv   = srcb_s[hcol];
      const float* tbase = tgt + (size_t)(b * Nn + j0 + wave * 16 + 8 * half) * Dd + hcol;
#pragma unroll
      for (int r = 0; r < 8; ++r) {
        const float tv  = tbase[(size_t)r * Dd];
        const float hv  = accC[r] + accD[r] + sv + tv;
        const float sig = 1.0f / (1.0f + __expf(-hv));
        partial[r] += (hv * sig) * w2v;
      }
    }
    // reduce partial over the 16 lanes spanning N=0..15
#pragma unroll
    for (int r = 0; r < 8; ++r) {
      float v = partial[r];
      v += __shfl_xor(v, 1, 32);
      v += __shfl_xor(v, 2, 32);
      v += __shfl_xor(v, 4, 32);
      v += __shfl_xor(v, 8, 32);
      if (l15 == 0)
        scores_s[j0 + wave * 16 + r + 8 * half] = v + b2v;
    }
    __syncthreads();
  }

  // ---- masked softmax over j ----
  float sc = (tid == i) ? -1.0e9f : scores_s[tid];
  float m = sc;
#pragma unroll
  for (int off = 16; off; off >>= 1) m = fmaxf(m, __shfl_xor(m, off, 32));
  if (lane == 0) red_s[wave] = m;
  __syncthreads();
  float mx = red_s[0];
#pragma unroll
  for (int k = 1; k < 8; ++k) mx = fmaxf(mx, red_s[k]);
  const float e = __expf(sc - mx);
  __syncthreads();                 // before reusing red_s
  float s = e;
#pragma unroll
  for (int off = 16; off; off >>= 1) s += __shfl_xor(s, off, 32);
  if (lane == 0) red_s[wave] = s;
  __syncthreads();
  float tot = red_s[0];
#pragma unroll
  for (int k = 1; k < 8; ++k) tot += red_s[k];
  out[bi * Nn + tid] = e / tot;
}

// ---------------------------------------------------------------------------
extern "C" void kernel_launch(void* const* d_in, const int* in_sizes, int n_in,
                              void* d_out, int out_size, void* d_ws, size_t ws_size,
                              hipStream_t stream) {
  const float* sh = (const float*)d_in[0];   // (B,N,D)
  const float* W1 = (const float*)d_in[1];   // (4D,D)
  const float* b1 = (const float*)d_in[2];   // (D)
  const float* W2 = (const float*)d_in[3];   // (D,1)
  const float* b2 = (const float*)d_in[4];   // (1)
  float* out = (float*)d_out;                // (B,N,N)

  char* ws = (char*)d_ws;
  float* srcb = (float*)ws;                                 // 1 MB
  float* tgt  = (float*)(ws + (1u << 20));                  // 1 MB
  unsigned short* WcS = (unsigned short*)(ws + (2u << 20));            // 128 KB
  unsigned short* WdS = (unsigned short*)(ws + (2u << 20) + (1u << 17)); // 128 KB

  sgb_prep<<<Bb * Nn + 128, 256, 0, stream>>>(sh, W1, b1, srcb, tgt, WcS, WdS);
  dim3 grid(Nn, Bb);
  sgb_main<<<grid, 256, 0, stream>>>(sh, W2, b2, srcb, tgt, WcS, WdS, out);
}
